// Sequence_79336635892484
// MI455X (gfx1250) — matrix operand order repaired and checked by reading, not compiled, not run on hardware
//
#include <hip/hip_runtime.h>

// ---------------------------------------------------------------------------
// 3-layer LSTM (H=51) + linear head, batch-parallel, weights LDS-resident,
// gate GEMMs via v_wmma_f32_16x16x32_bf16 (wave32). One wave per workgroup,
// 16 batch rows per wave, 64 workgroups for B=1024. Entire T+future sequence
// runs inside one kernel launch (recurrence is wave-local).
//
// Round-2 changes: GEMMs keep all 13 n-tile accumulators in registers and
// defer the gates-buffer stores until after the WMMA loop (so B-operand
// ds_loads overlap the matrix pipe instead of stalling on s_wait_dscnt 0x0
// caused by interleaved stores); dual-matmul layers use two independent
// 2-deep WMMA chains combined with VALU adds (co-executes with XDL pipe);
// output dot product reduced across all 32 lanes with a shuffle.
// ---------------------------------------------------------------------------

typedef __attribute__((ext_vector_type(16))) __bf16 v16bf;
typedef __attribute__((ext_vector_type(8)))  float  v8f;

#define HDIM 51
#define G4   204          // 4*H gate columns
#define GP   208          // padded gate columns (13 * 16)
#define NTI  13           // N tiles of 16
#define KTI  2            // K tiles of 32 (H=51 padded to 64)
#define MB   16           // batch rows per wave (WMMA M)
#define FRAG 512          // bf16 elements per 16x32 (A) / 32x16 (B) fragment

// ---- LDS layout (bytes); fragment regions are 32B aligned for v16bf loads
#define OFF_WFRAG   0
#define SZ_WFRAG    (5 * NTI * KTI * FRAG * 2)          // 133120
#define OFF_GATES   (OFF_WFRAG + SZ_WFRAG)              // 133120
#define SZ_GATES    (MB * GP * 4)                       // 13312
#define OFF_HFRAG   (OFF_GATES + SZ_GATES)              // 146432 (32B aligned)
#define SZ_HFRAG    (3 * KTI * FRAG * 2)                // 6144
#define OFF_CBUF    (OFF_HFRAG + SZ_HFRAG)              // 152576
#define SZ_CBUF     (3 * MB * 52 * 4)                   // 9984
#define OFF_H3ROW   (OFF_CBUF + SZ_CBUF)                // 162560
#define SZ_H3ROW    (MB * 52 * 4)                       // 3328
#define OFF_BIAS    (OFF_H3ROW + SZ_H3ROW)              // 165888
#define SZ_BIAS     (3 * G4 * 4)                        // 2448
#define OFF_WIH1    (OFF_BIAS + SZ_BIAS)                // 168336
#define SZ_WIH1     (G4 * 4)                            // 816
#define OFF_WLIN    (OFF_WIH1 + SZ_WIH1)                // 169152
#define SZ_WLIN     (52 * 4)                            // 208
#define OFF_XBUF    (OFF_WLIN + SZ_WLIN)                // 169360
#define SZ_XBUF     (MB * 4)                            // 64
#define OFF_BLIN    (OFF_XBUF + SZ_XBUF)                // 169424
#define LDS_BYTES   (OFF_BLIN + 32)                     // ~169.5 KB

__device__ __forceinline__ float sigm(float x) { return 1.0f / (1.0f + __expf(-x)); }

__global__ void __launch_bounds__(32, 1)
lstm3_wmma_kernel(const float* __restrict__ input,   // [B, T]
                  const float* __restrict__ Wih1,    // [4H, 1]
                  const float* __restrict__ Whh1,    // [4H, H]
                  const float* __restrict__ bih1, const float* __restrict__ bhh1,
                  const float* __restrict__ Wih2, const float* __restrict__ Whh2,
                  const float* __restrict__ bih2, const float* __restrict__ bhh2,
                  const float* __restrict__ Wih3, const float* __restrict__ Whh3,
                  const float* __restrict__ bih3, const float* __restrict__ bhh3,
                  const float* __restrict__ Wlin,    // [1, H]
                  const float* __restrict__ blin,    // [1]
                  const int*   __restrict__ futurep, // [1]
                  float* __restrict__ out,           // [B, T+future]
                  int T)
{
    extern __shared__ char smem[];
    __bf16* wfrag = (__bf16*)(smem + OFF_WFRAG);
    float*  gates = (float*) (smem + OFF_GATES);
    __bf16* hfrag = (__bf16*)(smem + OFF_HFRAG);
    float*  cbuf  = (float*) (smem + OFF_CBUF);
    float*  h3row = (float*) (smem + OFF_H3ROW);
    float*  biasL = (float*) (smem + OFF_BIAS);
    float*  wih1c = (float*) (smem + OFF_WIH1);
    float*  wlinc = (float*) (smem + OFF_WLIN);
    float*  xbuf  = (float*) (smem + OFF_XBUF);
    float*  blinc = (float*) (smem + OFF_BLIN);

    const int tid = threadIdx.x;

    // ---------------- one-time setup: state init + weight swizzle to LDS ----
    for (int i = tid; i < 3 * KTI * FRAG; i += 32) hfrag[i] = (__bf16)0.0f;   // h=0 (incl. K pad)
    for (int i = tid; i < 3 * MB * 52;    i += 32) cbuf[i]  = 0.0f;           // c=0
    for (int i = tid; i < G4; i += 32) {
        biasL[0 * G4 + i] = bih1[i] + bhh1[i];
        biasL[1 * G4 + i] = bih2[i] + bhh2[i];
        biasL[2 * G4 + i] = bih3[i] + bhh3[i];
        wih1c[i] = Wih1[i];                       // [4H,1] column
    }
    for (int i = tid; i < 52; i += 32) wlinc[i] = (i < HDIM) ? Wlin[i] : 0.0f;
    if (tid == 0) blinc[0] = blin[0];

    // B-fragment swizzle: W^T tile (kt,nt), element (k,n):
    //   lane = (n%16) + 16*(k>>4 within tile), halfword = k%16  (32x16 bf16 B layout)
    const float* wsrc[5] = { Whh1, Wih2, Whh2, Wih3, Whh3 };   // all [4H, H] row-major
    for (int m = 0; m < 5; ++m) {
        __bf16* dst = wfrag + m * NTI * KTI * FRAG;
        const float* W = wsrc[m];
        for (int idx = tid; idx < NTI * KTI * FRAG; idx += 32) {
            int tile   = idx >> 9;          // nt*KTI + kt
            int slot   = idx & 511;
            int lane_s = slot >> 4;
            int hw     = slot & 15;
            int nt = tile / KTI, kt = tile % KTI;
            int n  = nt * 16 + (lane_s & 15);
            int k  = kt * 32 + (lane_s >> 4) * 16 + hw;
            float v = (n < G4 && k < HDIM) ? W[n * HDIM + k] : 0.0f;  // W^T[k][n] = W[n][k]
            dst[idx] = (__bf16)v;
        }
    }
    __syncthreads();

    const int fut = futurep[0];
    const int TF  = T + fut;
    const int bg0 = blockIdx.x * MB;

    // D-fragment -> gates LDS store: VGPR r holds row (r + 8*(lane>>4)), col = nt*16 + (lane&15)
    auto store_acc = [&](int nt, const v8f& acc) {
        int n  = nt * 16 + (tid & 15);
        int mb = (tid >> 4) * 8;
#pragma unroll
        for (int r = 0; r < 8; ++r) gates[(mb + r) * GP + n] = acc[r];
    };

    // gates = h[hL] @ W[wM]^T   (layer 1: x handled as rank-1 add in EW).
    // All 13 accumulators live in registers; stores deferred so B-operand
    // ds_loads pipeline against the WMMAs.
    auto gemm1 = [&](int hL, int wM) {
        const v16bf* hv = (const v16bf*)(hfrag + hL * KTI * FRAG);
        const v16bf* wv = (const v16bf*)(wfrag + wM * NTI * KTI * FRAG);
        v16bf a0 = hv[tid], a1 = hv[32 + tid];
        v8f accs[NTI];
#pragma unroll
        for (int nt = 0; nt < NTI; ++nt) {
            v8f acc = {};
            acc = __builtin_amdgcn_wmma_f32_16x16x32_bf16(false, a0, false,
                      wv[(nt * 2 + 0) * 32 + tid], (short)0, acc, false, false);
            acc = __builtin_amdgcn_wmma_f32_16x16x32_bf16(false, a1, false,
                      wv[(nt * 2 + 1) * 32 + tid], (short)0, acc, false, false);
            accs[nt] = acc;
        }
#pragma unroll
        for (int nt = 0; nt < NTI; ++nt) store_acc(nt, accs[nt]);
    };

    // gates = h[hPrev] @ W[wIh]^T + h[hOwn] @ W[wHh]^T.
    // Two independent 2-deep WMMA chains per tile, combined with VALU adds
    // (VALU co-executes with the XDL matrix pipe); stores deferred.
    auto gemm2 = [&](int hPrev, int wIh, int hOwn, int wHh) {
        const v16bf* pv = (const v16bf*)(hfrag + hPrev * KTI * FRAG);
        const v16bf* ov = (const v16bf*)(hfrag + hOwn  * KTI * FRAG);
        const v16bf* iv = (const v16bf*)(wfrag + wIh * NTI * KTI * FRAG);
        const v16bf* hv = (const v16bf*)(wfrag + wHh * NTI * KTI * FRAG);
        v16bf p0 = pv[tid], p1 = pv[32 + tid];
        v16bf o0 = ov[tid], o1 = ov[32 + tid];
        v8f accs[NTI];
#pragma unroll
        for (int nt = 0; nt < NTI; ++nt) {
            v8f accA = {};
            v8f accB = {};
            accA = __builtin_amdgcn_wmma_f32_16x16x32_bf16(false, p0, false,
                       iv[(nt * 2 + 0) * 32 + tid], (short)0, accA, false, false);
            accB = __builtin_amdgcn_wmma_f32_16x16x32_bf16(false, o0, false,
                       hv[(nt * 2 + 0) * 32 + tid], (short)0, accB, false, false);
            accA = __builtin_amdgcn_wmma_f32_16x16x32_bf16(false, p1, false,
                       iv[(nt * 2 + 1) * 32 + tid], (short)0, accA, false, false);
            accB = __builtin_amdgcn_wmma_f32_16x16x32_bf16(false, o1, false,
                       hv[(nt * 2 + 1) * 32 + tid], (short)0, accB, false, false);
            accs[nt] = accA + accB;
        }
#pragma unroll
        for (int nt = 0; nt < NTI; ++nt) store_acc(nt, accs[nt]);
    };

    // elementwise: gate nonlinearities + cell update; writes h (bf16) straight
    // into the 16x32 A-fragment layout: lane = m + 16*(k>>4), hw = k&15
    auto ew = [&](int L, bool addx, bool cap3) {
        const float* bia = biasL + L * G4;
        float* cb  = cbuf + L * MB * 52;
        __bf16* hf = hfrag + L * KTI * FRAG;
        for (int idx = tid; idx < MB * HDIM; idx += 32) {
            int b = idx / HDIM;
            int j = idx - b * HDIM;
            float gi = gates[b * GP + j]            + bia[j];
            float gf = gates[b * GP + HDIM + j]     + bia[HDIM + j];
            float gg = gates[b * GP + 2 * HDIM + j] + bia[2 * HDIM + j];
            float go = gates[b * GP + 3 * HDIM + j] + bia[3 * HDIM + j];
            if (addx) {   // layer-1 scalar input: rank-1 x_b * W_ih1 column
                float xb = xbuf[b];
                gi += xb * wih1c[j];
                gf += xb * wih1c[HDIM + j];
                gg += xb * wih1c[2 * HDIM + j];
                go += xb * wih1c[3 * HDIM + j];
            }
            float c  = cb[b * 52 + j];
            float cn = sigm(gf) * c + sigm(gi) * tanhf(gg);
            float hn = sigm(go) * tanhf(cn);
            cb[b * 52 + j] = cn;
            int kt = j >> 5, kk = j & 31;
            int lane_s = b + 16 * (kk >> 4);
            int hw     = kk & 15;
            hf[kt * FRAG + lane_s * 16 + hw] = (__bf16)hn;
            if (cap3) h3row[b * 52 + j] = hn;
        }
    };

    // ------------------------------- time loop ------------------------------
    for (int t = 0; t < TF; ++t) {
        if (t < T && tid < MB)
            xbuf[tid] = input[(size_t)(bg0 + tid) * T + t];
        // (t >= T: xbuf already holds previous step's output)

        gemm1(0, 0);                 // h1 @ Whh1^T
        __syncthreads();
        ew(0, true, false);
        __syncthreads();
        gemm2(0, 1, 1, 2);           // h1 @ Wih2^T + h2 @ Whh2^T
        __syncthreads();
        ew(1, false, false);
        __syncthreads();
        gemm2(1, 3, 2, 4);           // h2 @ Wih3^T + h3 @ Whh3^T
        __syncthreads();
        ew(2, false, true);
        __syncthreads();

        // linear head: out = h3 . Wlin + blin, split across both lane halves
        {
            int b    = tid & 15;
            int half = tid >> 4;            // lanes 0-15: j in [0,26), 16-31: [26,51)
            int j0   = half * 26;
            int j1   = half ? HDIM : 26;
            float d = 0.0f;
            for (int j = j0; j < j1; ++j) d += h3row[b * 52 + j] * wlinc[j];
            d += __shfl_xor(d, 16, 32);     // combine the two halves
            if (tid < MB) {
                d += blinc[0];
                out[(size_t)(bg0 + b) * TF + t] = d;
                xbuf[b] = d;                // autoregressive feedback for t >= T
            }
        }
        __syncthreads();
    }
}

extern "C" void kernel_launch(void* const* d_in, const int* in_sizes, int n_in,
                              void* d_out, int out_size, void* d_ws, size_t ws_size,
                              hipStream_t stream) {
    (void)n_in; (void)out_size; (void)d_ws; (void)ws_size;
    const float* input = (const float*)d_in[0];
    const float* Wih1  = (const float*)d_in[1];
    const float* Whh1  = (const float*)d_in[2];
    const float* bih1  = (const float*)d_in[3];
    const float* bhh1  = (const float*)d_in[4];
    const float* Wih2  = (const float*)d_in[5];
    const float* Whh2  = (const float*)d_in[6];
    const float* bih2  = (const float*)d_in[7];
    const float* bhh2  = (const float*)d_in[8];
    const float* Wih3  = (const float*)d_in[9];
    const float* Whh3  = (const float*)d_in[10];
    const float* bih3  = (const float*)d_in[11];
    const float* bhh3  = (const float*)d_in[12];
    const float* Wlin  = (const float*)d_in[13];
    const float* blin  = (const float*)d_in[14];
    const int*   futp  = (const int*)  d_in[15];

    const int B = 1024;                 // reference batch
    const int T = in_sizes[0] / B;      // input is [B, T]

    (void)hipFuncSetAttribute((const void*)lstm3_wmma_kernel,
                              hipFuncAttributeMaxDynamicSharedMemorySize,
                              (int)LDS_BYTES);

    lstm3_wmma_kernel<<<B / MB, 32, LDS_BYTES, stream>>>(
        input, Wih1, Whh1, bih1, bhh1,
        Wih2, Whh2, bih2, bhh2,
        Wih3, Whh3, bih3, bhh3,
        Wlin, blin, futp, (float*)d_out, T);
}
